// MultiheadSelfAttention_2113123910291
// MI455X (gfx1250) — matrix operand order repaired and checked by reading, not compiled, not run on hardware
//
#include <hip/hip_runtime.h>
#include <math.h>

// ---------------------------------------------------------------------------
// MI455X (gfx1250) multi-head self-attention with RoPE, causal, f16 WMMA.
//   K0: f32 -> f16 conversion of x and the four weight matrices
//   K1: fused QKV projection + RoPE. 4-wave WG; weight tile double-buffered in
//       LDS via global_load_async_to_lds_b128 (s_wait_asynccnt + barriers).
//   K2: flash attention per (b,h,16-query tile): QK^T (WMMA), online softmax,
//       P transposed through LDS, P@V (WMMA).
//   K3: output projection O @ Wo^T, same async-LDS staging -> f32 d_out.
// ---------------------------------------------------------------------------

typedef __attribute__((ext_vector_type(16))) _Float16 v16h;
typedef __attribute__((ext_vector_type(8)))  _Float16 v8h;
typedef __attribute__((ext_vector_type(8)))  float    v8f;
typedef int v4i_vs __attribute__((vector_size(16)));   // matches builtin pointee

#define D_MODEL   2048
#define NUM_HEADS 16
#define D_K       128
#define SEQ       2048
#define BATCH     4
#define MROWS     (BATCH * SEQ)   // 8192 flattened (b,s) rows
#define KTILE     32
#define NKT       (D_MODEL / KTILE)   // 64 k-steps

// ---- gfx1250 async global->LDS copy, with graceful fallback ---------------
#if defined(__has_builtin)
#  if __has_builtin(__builtin_amdgcn_global_load_async_to_lds_b128)
#    define HAVE_ASYNC_LDS 1
#  endif
#endif
#ifndef HAVE_ASYNC_LDS
#  define HAVE_ASYNC_LDS 0
#endif

#if HAVE_ASYNC_LDS
#  if __has_builtin(__builtin_amdgcn_s_wait_asynccnt)
#    define WAIT_ASYNC(n) __builtin_amdgcn_s_wait_asynccnt(n)
#  else
#    define WAIT_ASYNC(n) asm volatile("s_wait_asynccnt %0" ::"n"(n) : "memory")
#  endif
#else
#  define WAIT_ASYNC(n) ((void)0)
#endif

__device__ __forceinline__ void async_cp16(const _Float16* g, _Float16* l) {
#if HAVE_ASYNC_LDS
    __builtin_amdgcn_global_load_async_to_lds_b128(
        (__attribute__((address_space(1))) v4i_vs*)g,
        (__attribute__((address_space(3))) v4i_vs*)l, 0, 0);
#else
    *(v8h*)l = *(const v8h*)g;   // synchronous fallback (16B)
#endif
}

// A-operand (16x32 f16) per-lane layout: lanes 0-15 hold row M=lane with
// K [k..k+16); lanes 16-31 hold [k+16..k+32)
__device__ __forceinline__ v16h load_a_frag(const _Float16* rowbase, int k, int lane) {
    const int off = k + ((lane >> 4) << 4);
    v8h lo = *(const v8h*)(rowbase + off);
    v8h hi = *(const v8h*)(rowbase + off + 8);
    v16h r;
#pragma unroll
    for (int i = 0; i < 8; ++i) { r[i] = lo[i]; r[i + 8] = hi[i]; }
    return r;
}

// B-operand (32x16 f16): lane holds column N=lane&15, 16 contiguous K values
// starting at k + 16*(lane>>4). For y=x@W^T this is simply a row of W.
__device__ __forceinline__ v16h load_b_frag(const _Float16* rowbase, int k, int lane) {
    return *(const v16h*)(rowbase + k + ((lane >> 4) << 4));
}

__device__ __forceinline__ v8f wmma_f16(v16h a, v16h b, v8f c) {
    return __builtin_amdgcn_wmma_f32_16x16x32_f16(
        /*neg_a=*/false, a, /*neg_b=*/false, b,
        /*c_mod=*/(short)0, c, /*reuse_a=*/false, /*reuse_b=*/false);
}

// ---------------------------------------------------------------------------
// K0: elementwise f32 -> f16
// ---------------------------------------------------------------------------
__global__ void __launch_bounds__(256)
cvt_f32_to_f16_kernel(const float* __restrict__ src, _Float16* __restrict__ dst, int n) {
    int i = blockIdx.x * blockDim.x + threadIdx.x;
    const int stride = gridDim.x * blockDim.x;
    for (; i < n; i += stride) dst[i] = (_Float16)src[i];
}

// ---------------------------------------------------------------------------
// K1: QKV projection + RoPE.
// grid = (NUM_HEADS, MROWS/128, 3), block = 128 (4 waves).
// WG output tile: 128 rows x 128 cols (one head). Wave w owns rows
// [w*32, w*32+32) as two 16-row WMMA tiles. The 128x32 weight tile for each
// k-step is staged in LDS (double buffered, async) and shared by all 4 waves.
// ---------------------------------------------------------------------------
__global__ void __launch_bounds__(128)
qkv_rope_kernel(const _Float16* __restrict__ xh,
                const _Float16* __restrict__ wq,
                const _Float16* __restrict__ wk,
                const _Float16* __restrict__ wv,
                const int*      __restrict__ tokpos,
                _Float16* __restrict__ qbuf,
                _Float16* __restrict__ kbuf,
                _Float16* __restrict__ vbuf) {
    const int tid   = threadIdx.x;
    const int lane  = tid & 31;
    const int wv_id = tid >> 5;          // wave in WG
    const int h     = blockIdx.x;        // head (128-wide output tile)
    const int m0    = blockIdx.y * 128;  // row tile over flattened (b,s)
    const int which = blockIdx.z;        // 0=Q, 1=K, 2=V
    const _Float16* W = (which == 0) ? wq : (which == 1) ? wk : wv;

    const int b  = m0 >> 11;             // m0 / SEQ (128-row tiles never cross batch)
    const int s0 = m0 & (SEQ - 1);

    __shared__ __align__(32) _Float16 lbs[2][128 * KTILE];   // 2 x 8KB weight tiles

    // staging map: thread t copies W row (h*128 + t), 64B = 4 x 16B chunks
    const _Float16* wsrc = W + (size_t)(h * D_K + tid) * D_MODEL;

    v8f c[2][8] = {};
    const _Float16* xrow0 = xh + (size_t)(m0 + wv_id * 32 + (lane & 15)) * D_MODEL;
    const _Float16* xrow1 = xrow0 + (size_t)16 * D_MODEL;

    // prologue: stage tile 0
#pragma unroll
    for (int cch = 0; cch < 4; ++cch)
        async_cp16(wsrc + cch * 8, &lbs[0][tid * KTILE] + cch * 8);

    for (int kt = 0; kt < NKT; ++kt) {
        const int k = kt * KTILE;
        if (kt + 1 < NKT) {
            const int nb = (kt + 1) & 1;
            const _Float16* gs = wsrc + (kt + 1) * KTILE;
#pragma unroll
            for (int cch = 0; cch < 4; ++cch)
                async_cp16(gs + cch * 8, &lbs[nb][tid * KTILE] + cch * 8);
            WAIT_ASYNC(4);               // tile kt complete, tile kt+1 in flight
        } else {
            WAIT_ASYNC(0);
        }
        __syncthreads();                 // all waves' tile kt visible in LDS

        __builtin_prefetch(xrow0 + k + 128, 0, 1);
        v16h a0 = load_a_frag(xrow0, k, lane);
        v16h a1 = load_a_frag(xrow1, k, lane);
        const _Float16* bbase = &lbs[kt & 1][0];
#pragma unroll
        for (int nn = 0; nn < 8; ++nn) {
            v16h bf = *(const v16h*)(bbase + (nn * 16 + (lane & 15)) * KTILE + ((lane >> 4) << 4));
            c[0][nn] = wmma_f16(a0, bf, c[0][nn]);
            c[1][nn] = wmma_f16(a1, bf, c[1][nn]);
        }
        __syncthreads();                 // readers done before buffer is overwritten
    }

    // Epilogue: C/D layout element (row = j + 8*(lane>>4), col = nn*16 + lane&15).
    const int rlo = (lane >> 4) << 3;
#pragma unroll
    for (int mt = 0; mt < 2; ++mt) {
        const int rbase = wv_id * 32 + mt * 16;
#pragma unroll
        for (int nn = 0; nn < 8; ++nn) {
#pragma unroll
            for (int j = 0; j < 8; ++j) {
                const int row = rbase + j + rlo;
                const int col = nn * 16 + (lane & 15);
                const int s   = s0 + row;
                float v = c[mt][nn][j];
                if (which == 2) {
                    // V: store transposed [b,h,d,s]
                    vbuf[((size_t)(b * NUM_HEADS + h) * D_K + col) * SEQ + s] = (_Float16)v;
                } else {
                    // RoPE: pair (2p,2p+1) on adjacent lanes -> shfl_xor(1)
                    float pv = __shfl_xor(v, 1, 32);
                    int pos = tokpos[s];
                    float freq = __expf(-(float)(col >> 1) * (2.0f / (float)D_K) * 9.210340371976184f);
                    float ang = (float)pos * freq;
                    float sv, cv;
                    __sincosf(ang, &sv, &cv);
                    float r = ((lane & 1) == 0) ? (cv * v - sv * pv)
                                                : (sv * pv + cv * v);
                    _Float16* dst = (which == 0) ? qbuf : kbuf;
                    dst[((size_t)(b * NUM_HEADS + h) * SEQ + s) * D_K + col] = (_Float16)r;
                }
            }
        }
    }
}

// ---------------------------------------------------------------------------
// K2: causal flash attention. grid = (SEQ/16, NUM_HEADS, BATCH), block = 32.
// ---------------------------------------------------------------------------
__global__ void __launch_bounds__(32)
attention_kernel(const _Float16* __restrict__ qbuf,
                 const _Float16* __restrict__ kbuf,
                 const _Float16* __restrict__ vbuf,
                 _Float16* __restrict__ obuf) {
    const int lane = threadIdx.x;
    const int q0   = blockIdx.x * 16;
    const int h    = blockIdx.y;
    const int b    = blockIdx.z;
    const size_t bh = (size_t)(b * NUM_HEADS + h);

    __shared__ __align__(16) _Float16 pl[16 * 32];   // P tile transpose buffer

    const _Float16* qrow = qbuf + (bh * SEQ + q0 + (lane & 15)) * D_K;
    v16h aq[4];
#pragma unroll
    for (int kk = 0; kk < 4; ++kk) aq[kk] = load_a_frag(qrow, kk * 32, lane);

    v8f acc[8] = {};
    float mrow[8], lrow[8];
#pragma unroll
    for (int j = 0; j < 8; ++j) { mrow[j] = -1e30f; lrow[j] = 0.0f; }

    const int rlo = (lane >> 4) << 3;
    const float scale = 0.08838834764831845f;        // 1/sqrt(128)
    const int nkb = ((q0 + 15) >> 5) + 1;            // causal key-block count

    for (int kb = 0; kb < nkb; ++kb) {
        const int kbase = kb * 32;

        v8f sc[2] = {};
#pragma unroll
        for (int t = 0; t < 2; ++t) {
            const _Float16* krow = kbuf + (bh * SEQ + kbase + t * 16 + (lane & 15)) * D_K;
#pragma unroll
            for (int kk = 0; kk < 4; ++kk) {
                v16h bk = load_b_frag(krow, kk * 32, lane);
                sc[t] = wmma_f16(aq[kk], bk, sc[t]);
            }
        }

#pragma unroll
        for (int j = 0; j < 8; ++j) {
            const int row = q0 + j + rlo;
            const int c0 = kbase + (lane & 15);
            const int c1 = c0 + 16;
            float s0v = sc[0][j] * scale; if (c0 > row) s0v = -1e30f;
            float s1v = sc[1][j] * scale; if (c1 > row) s1v = -1e30f;

            float pm = fmaxf(s0v, s1v);
#pragma unroll
            for (int off = 1; off < 16; off <<= 1) pm = fmaxf(pm, __shfl_xor(pm, off, 32));
            const float mnew  = fmaxf(mrow[j], pm);
            const float alpha = __expf(mrow[j] - mnew);
            const float p0 = __expf(s0v - mnew);
            const float p1 = __expf(s1v - mnew);
            float psum = p0 + p1;
#pragma unroll
            for (int off = 1; off < 16; off <<= 1) psum += __shfl_xor(psum, off, 32);
            lrow[j] = lrow[j] * alpha + psum;
            mrow[j] = mnew;
#pragma unroll
            for (int n = 0; n < 8; ++n) acc[n][j] *= alpha;

            const int lr = j + rlo;
            pl[lr * 32 + (lane & 15)]      = (_Float16)p0;
            pl[lr * 32 + 16 + (lane & 15)] = (_Float16)p1;
        }
        __syncthreads();

        v16h pa;
        {
            const _Float16* prb = pl + (lane & 15) * 32 + ((lane >> 4) << 4);
            v8h lo = *(const v8h*)(prb);
            v8h hi = *(const v8h*)(prb + 8);
#pragma unroll
            for (int i = 0; i < 8; ++i) { pa[i] = lo[i]; pa[i + 8] = hi[i]; }
        }

#pragma unroll
        for (int n = 0; n < 8; ++n) {
            const _Float16* vrow = vbuf + (bh * D_K + n * 16 + (lane & 15)) * SEQ + kbase;
            v16h bvf = *(const v16h*)(vrow + ((lane >> 4) << 4));
            acc[n] = wmma_f16(pa, bvf, acc[n]);
        }
        __syncthreads();   // protect pl WAR for next block
    }

#pragma unroll
    for (int j = 0; j < 8; ++j) {
        const float inv = 1.0f / lrow[j];
        const int row = q0 + j + rlo;
        _Float16* orow = obuf + ((size_t)b * SEQ + row) * D_MODEL + h * D_K;
#pragma unroll
        for (int n = 0; n < 8; ++n)
            orow[n * 16 + (lane & 15)] = (_Float16)(acc[n][j] * inv);
    }
}

// ---------------------------------------------------------------------------
// K3: out = O @ Wo^T -> f32. grid = (D_MODEL/128, MROWS/128), block = 128.
// Same async-LDS double-buffered weight staging as K1.
// ---------------------------------------------------------------------------
__global__ void __launch_bounds__(128)
out_proj_kernel(const _Float16* __restrict__ oh,
                const _Float16* __restrict__ wo,
                float* __restrict__ out) {
    const int tid   = threadIdx.x;
    const int lane  = tid & 31;
    const int wv_id = tid >> 5;
    const int n0 = blockIdx.x * 128;
    const int m0 = blockIdx.y * 128;

    __shared__ __align__(32) _Float16 lbs[2][128 * KTILE];

    const _Float16* wsrc = wo + (size_t)(n0 + tid) * D_MODEL;

    v8f c[2][8] = {};
    const _Float16* orow0 = oh + (size_t)(m0 + wv_id * 32 + (lane & 15)) * D_MODEL;
    const _Float16* orow1 = orow0 + (size_t)16 * D_MODEL;

#pragma unroll
    for (int cch = 0; cch < 4; ++cch)
        async_cp16(wsrc + cch * 8, &lbs[0][tid * KTILE] + cch * 8);

    for (int kt = 0; kt < NKT; ++kt) {
        const int k = kt * KTILE;
        if (kt + 1 < NKT) {
            const int nb = (kt + 1) & 1;
            const _Float16* gs = wsrc + (kt + 1) * KTILE;
#pragma unroll
            for (int cch = 0; cch < 4; ++cch)
                async_cp16(gs + cch * 8, &lbs[nb][tid * KTILE] + cch * 8);
            WAIT_ASYNC(4);
        } else {
            WAIT_ASYNC(0);
        }
        __syncthreads();

        __builtin_prefetch(orow0 + k + 128, 0, 1);
        v16h a0 = load_a_frag(orow0, k, lane);
        v16h a1 = load_a_frag(orow1, k, lane);
        const _Float16* bbase = &lbs[kt & 1][0];
#pragma unroll
        for (int nn = 0; nn < 8; ++nn) {
            v16h bf = *(const v16h*)(bbase + (nn * 16 + (lane & 15)) * KTILE + ((lane >> 4) << 4));
            c[0][nn] = wmma_f16(a0, bf, c[0][nn]);
            c[1][nn] = wmma_f16(a1, bf, c[1][nn]);
        }
        __syncthreads();
    }

    const int rlo = (lane >> 4) << 3;
#pragma unroll
    for (int mt = 0; mt < 2; ++mt) {
#pragma unroll
        for (int j = 0; j < 8; ++j) {
            const size_t row = (size_t)(m0 + wv_id * 32 + mt * 16 + j + rlo);
#pragma unroll
            for (int nn = 0; nn < 8; ++nn)
                out[row * D_MODEL + n0 + nn * 16 + (lane & 15)] = c[mt][nn][j];
        }
    }
}

// ---------------------------------------------------------------------------
extern "C" void kernel_launch(void* const* d_in, const int* in_sizes, int n_in,
                              void* d_out, int out_size, void* d_ws, size_t ws_size,
                              hipStream_t stream) {
    const float* x   = (const float*)d_in[0];
    const int*   tok = (const int*)d_in[1];
    const float* Wq  = (const float*)d_in[2];
    const float* Wk  = (const float*)d_in[3];
    const float* Wv  = (const float*)d_in[4];
    const float* Wo  = (const float*)d_in[5];
    float* out = (float*)d_out;

    _Float16* ws = (_Float16*)d_ws;
    size_t off = 0;
    _Float16* xh  = ws + off; off += (size_t)MROWS * D_MODEL;
    _Float16* wqh = ws + off; off += (size_t)D_MODEL * D_MODEL;
    _Float16* wkh = ws + off; off += (size_t)D_MODEL * D_MODEL;
    _Float16* wvh = ws + off; off += (size_t)D_MODEL * D_MODEL;
    _Float16* woh = ws + off; off += (size_t)D_MODEL * D_MODEL;
    _Float16* qb  = ws + off; off += (size_t)MROWS * D_MODEL;   // [b,h,s,d]
    _Float16* kb  = ws + off; off += (size_t)MROWS * D_MODEL;   // [b,h,s,d]
    _Float16* vb  = ws + off; off += (size_t)MROWS * D_MODEL;   // [b,h,d,s]
    _Float16* ob  = ws + off; off += (size_t)MROWS * D_MODEL;   // [b,s,e]

    const int nx = MROWS * D_MODEL;
    const int nw = D_MODEL * D_MODEL;
    cvt_f32_to_f16_kernel<<<2048, 256, 0, stream>>>(x,  xh,  nx);
    cvt_f32_to_f16_kernel<<<1024, 256, 0, stream>>>(Wq, wqh, nw);
    cvt_f32_to_f16_kernel<<<1024, 256, 0, stream>>>(Wk, wkh, nw);
    cvt_f32_to_f16_kernel<<<1024, 256, 0, stream>>>(Wv, wvh, nw);
    cvt_f32_to_f16_kernel<<<1024, 256, 0, stream>>>(Wo, woh, nw);

    dim3 g1(NUM_HEADS, MROWS / 128, 3);
    qkv_rope_kernel<<<g1, 128, 0, stream>>>(xh, wqh, wkh, wvh, tok, qb, kb, vb);

    dim3 g2(SEQ / 16, NUM_HEADS, BATCH);
    attention_kernel<<<g2, 32, 0, stream>>>(qb, kb, vb, ob);

    dim3 g3(D_MODEL / 128, MROWS / 128);
    out_proj_kernel<<<g3, 128, 0, stream>>>(ob, woh, out);
}
